// MixtralAttention_53223234732073
// MI455X (gfx1250) — matrix operand order repaired and checked by reading, not compiled, not run on hardware
//
#include <hip/hip_runtime.h>
#include <hip/hip_fp16.h>

typedef __attribute__((ext_vector_type(16))) _Float16 v16h;
typedef __attribute__((ext_vector_type(4)))  _Float16 v4h;
typedef __attribute__((ext_vector_type(8)))  float    v8f;
typedef _Float16 half_t;

static constexpr int S    = 2048;
static constexpr int D    = 4096;
static constexpr int H    = 32;
static constexpr int KVH  = 8;
static constexpr int HD   = 128;
static constexpr int QKVN = H * HD + 2 * KVH * HD;   // 6144
static constexpr float SCALE = 0.08838834764831845f; // 128^-0.5

union Frag16 { v16h h; uint4 u[2]; };

__device__ __forceinline__ v8f v8f_zero() {
  v8f z;
#pragma unroll
  for (int i = 0; i < 8; ++i) z[i] = 0.0f;
  return z;
}

__device__ __forceinline__ v8f wmma_f16(v16h a, v16h b, v8f c) {
  // D = A(16x32 f16) * B(32x16 f16) + C(16x16 f32)
  return __builtin_amdgcn_wmma_f32_16x16x32_f16(
      /*neg_a=*/false, a, /*neg_b=*/false, b,
      /*c_mod=*/(short)0, c, /*reuse_a=*/false, /*reuse_b=*/false);
}

// ---------------------------------------------------------------------------
// Cast fp32 -> f16 (vectorized)
// ---------------------------------------------------------------------------
__global__ __launch_bounds__(256) void cast_f32_f16_kernel(
    const float* __restrict__ src, half_t* __restrict__ dst, long n) {
  long i = ((long)blockIdx.x * 256 + threadIdx.x) * 4;
  if (i + 3 < n) {
    float4 v = *(const float4*)(src + i);
    v4h o;
    o[0] = (half_t)v.x; o[1] = (half_t)v.y;
    o[2] = (half_t)v.z; o[3] = (half_t)v.w;
    *(v4h*)(dst + i) = o;
  }
}

// ---------------------------------------------------------------------------
// Transpose + cast: W[K][N] fp32 -> Wt[N][K] f16  (32x32 LDS tiles)
// ---------------------------------------------------------------------------
__global__ __launch_bounds__(256) void transpose_cast_kernel(
    const float* __restrict__ W, half_t* __restrict__ Wt, int K, int Ncols) {
  __shared__ float tile[32][33];
  const int bx = blockIdx.x * 32;  // along N
  const int by = blockIdx.y * 32;  // along K
  const int x = threadIdx.x;       // 0..31
  const int y = threadIdx.y;       // 0..7
#pragma unroll
  for (int p = 0; p < 32; p += 8)
    tile[y + p][x] = W[(long)(by + y + p) * Ncols + bx + x];
  __syncthreads();
#pragma unroll
  for (int p = 0; p < 32; p += 8)
    Wt[(long)(bx + y + p) * K + by + x] = (half_t)tile[x][y + p];
}

// ---------------------------------------------------------------------------
// NT GEMM: C[M][N] = A[M][K](f16) * Bt[N][K](f16)^T, f32 accumulate.
// Block tile 128x128, 8 waves (4x2), wave tile 32x64, K-step 32.
// ---------------------------------------------------------------------------
template <typename OutT>
__global__ __launch_bounds__(256) void gemm_nt_kernel(
    const half_t* __restrict__ A, const half_t* __restrict__ Bt,
    OutT* __restrict__ C, int M, int N, int K) {
  __shared__ half_t lA[128 * 32];
  __shared__ half_t lB[128 * 32];

  const int tid  = threadIdx.x;
  const int lane = tid & 31;
  const int wave = tid >> 5;
  const int wm   = wave >> 1;   // 0..3
  const int wn   = wave & 1;    // 0..1
  const long bM  = (long)blockIdx.x * 128;
  const long bN  = (long)blockIdx.y * 128;

  v8f acc[2][4];
#pragma unroll
  for (int mi = 0; mi < 2; ++mi)
#pragma unroll
    for (int ni = 0; ni < 4; ++ni) acc[mi][ni] = v8f_zero();

  const int lrow  = tid >> 1;   // 0..127
  const int lhalf = tid & 1;    // which 16-f16 half of the 32-wide K slab
  const half_t* gA = A  + (bM + lrow) * (long)K + lhalf * 16;
  const half_t* gB = Bt + (bN + lrow) * (long)K + lhalf * 16;
  uint4* sA = (uint4*)(lA + lrow * 32 + lhalf * 16);
  uint4* sB = (uint4*)(lB + lrow * 32 + lhalf * 16);

  const int kbA  = (lane >> 4) * 8;    // A-fragment interleaved K base
  const int kbB  = (lane >> 4) * 8;    // B-fragment K base
  const int mrow = lane & 15;

  for (int k0 = 0; k0 < K; k0 += 32) {
    uint4 a0 = *(const uint4*)(gA + k0);
    uint4 a1 = *(const uint4*)(gA + k0 + 8);
    uint4 b0 = *(const uint4*)(gB + k0);
    uint4 b1 = *(const uint4*)(gB + k0 + 8);
    __syncthreads();
    sA[0] = a0; sA[1] = a1;
    sB[0] = b0; sB[1] = b1;
    __syncthreads();
    if (k0 + 32 < K) {
      __builtin_prefetch(gA + k0 + 32, 0, 1);
      __builtin_prefetch(gB + k0 + 32, 0, 1);
    }
    v16h af[2], bf[4];
#pragma unroll
    for (int mi = 0; mi < 2; ++mi) {
      const half_t* p = lA + (wm * 32 + mi * 16 + mrow) * 32;
      Frag16 f;
      f.u[0] = *(const uint4*)(p + kbA);
      f.u[1] = *(const uint4*)(p + 16 + kbA);
      af[mi] = f.h;
    }
#pragma unroll
    for (int ni = 0; ni < 4; ++ni) {
      const half_t* p = lB + (wn * 64 + ni * 16 + mrow) * 32 + kbB;
      Frag16 f;
      f.u[0] = *(const uint4*)(p);
      f.u[1] = *(const uint4*)(p + 16);
      bf[ni] = f.h;
    }
#pragma unroll
    for (int mi = 0; mi < 2; ++mi)
#pragma unroll
      for (int ni = 0; ni < 4; ++ni)
        acc[mi][ni] = wmma_f16(af[mi], bf[ni], acc[mi][ni]);
  }

  const int rb = (lane >> 4) * 8;  // row base within 16x16 D tile
  const int cn = lane & 15;
#pragma unroll
  for (int mi = 0; mi < 2; ++mi)
#pragma unroll
    for (int ni = 0; ni < 4; ++ni)
#pragma unroll
      for (int j = 0; j < 8; ++j)
        C[(bM + wm * 32 + mi * 16 + rb + j) * (long)N +
          bN + wn * 64 + ni * 16 + cn] = (OutT)acc[mi][ni][j];
}

// ---------------------------------------------------------------------------
// RoPE + relayout: qkv[S][6144] f16 -> Qh[H][S][HD] (pre-scaled), Kh/Vh[KVH][S][HD]
// ---------------------------------------------------------------------------
__global__ __launch_bounds__(128) void rope_relayout_kernel(
    const half_t* __restrict__ qkv, const float* __restrict__ cosb,
    const float* __restrict__ sinb, half_t* __restrict__ Qh,
    half_t* __restrict__ Kh, half_t* __restrict__ Vh) {
  const int s = blockIdx.x;
  const int d = threadIdx.x;                // 0..127
  const float c  = cosb[s * HD + d];
  const float sn = sinb[s * HD + d];
  const float sg = (d < 64) ? -1.0f : 1.0f; // rotate_half sign
  const int dp = (d + 64) & 127;
  const half_t* row = qkv + (long)s * QKVN;
#pragma unroll 4
  for (int h = 0; h < H; ++h) {
    const half_t* x = row + h * HD;
    float r = (float)x[d] * c + sg * (float)x[dp] * sn;
    Qh[((long)h * S + s) * HD + d] = (half_t)(r * SCALE);
  }
#pragma unroll 4
  for (int h = 0; h < KVH; ++h) {
    const half_t* x = row + D + h * HD;
    float r = (float)x[d] * c + sg * (float)x[dp] * sn;
    Kh[((long)h * S + s) * HD + d] = (half_t)r;
    Vh[((long)h * S + s) * HD + d] = row[D + KVH * HD + h * HD + d];
  }
}

// ---------------------------------------------------------------------------
// Flash attention: grid (S/64, H), 128 threads = 4 waves x 16 query rows.
// Key chunks of 32; K tile [32][HD] row-major, V tile [HD][32] transposed.
// ---------------------------------------------------------------------------
__global__ __launch_bounds__(128) void flash_attn_kernel(
    const half_t* __restrict__ Qh, const half_t* __restrict__ Kh,
    const half_t* __restrict__ Vh, const unsigned char* __restrict__ amask,
    half_t* __restrict__ O) {
  __shared__ half_t lK[32 * HD];      // [key][hd]
  __shared__ half_t lV[HD * 32];      // [hd][key] (transposed)
  __shared__ half_t lP[4 * 16 * 32];  // per-wave P staging

  const int qt   = blockIdx.x;
  const int h    = blockIdx.y;
  const int kvh  = h >> 2;            // GROUPS = 4
  const int tid  = threadIdx.x;
  const int lane = tid & 31;
  const int wave = tid >> 5;
  const int q0   = qt * 64 + wave * 16;

  const int col = lane & 15;
  const int kbA = (lane >> 4) * 8;
  const int kbB = (lane >> 4) * 8;
  const int rb  = (lane >> 4) * 8;

  // Q fragments (A-layout), 4 x K32 slabs over HD=128
  v16h qf[4];
  {
    const half_t* qp = Qh + ((long)h * S + q0 + col) * HD;
#pragma unroll
    for (int t = 0; t < 4; ++t) {
      Frag16 f;
      f.u[0] = *(const uint4*)(qp + t * 32 + kbA);
      f.u[1] = *(const uint4*)(qp + t * 32 + 16 + kbA);
      qf[t] = f.h;
    }
  }

  v8f oacc[8];
#pragma unroll
  for (int t = 0; t < 8; ++t) oacc[t] = v8f_zero();
  float mrun[8], lrun[8];
#pragma unroll
  for (int j = 0; j < 8; ++j) { mrun[j] = -1e30f; lrun[j] = 0.0f; }

  const int krow = tid >> 2;          // cooperative K-tile row 0..31
  const int kseg = (tid & 3) * 32;    // halfword offset (64B segment)
  const int kend = qt * 64 + 64;      // causal bound for this block

  for (int kc = 0; kc < kend; kc += 32) {
    __syncthreads();
    {  // K tile, row-major
      const uint4* g = (const uint4*)(Kh + ((long)kvh * S + kc + krow) * HD + kseg);
      uint4* sdst = (uint4*)(lK + krow * HD + kseg);
#pragma unroll
      for (int i = 0; i < 4; ++i) sdst[i] = g[i];
    }
    {  // V tile, transposed: thread <-> hd column; 64B/wave coalesced reads
      half_t vv[32];
      const half_t* g = Vh + ((long)kvh * S + kc) * HD + tid;
#pragma unroll
      for (int k = 0; k < 32; ++k) vv[k] = g[(long)k * HD];
      uint4* sdst = (uint4*)(lV + tid * 32);
      const uint4* src = (const uint4*)vv;
#pragma unroll
      for (int i = 0; i < 4; ++i) sdst[i] = src[i];
    }
    __syncthreads();

    // scores: two 16x16 tiles (keys kc..kc+15, kc+16..kc+31)
    v8f s0 = v8f_zero(), s1 = v8f_zero();
#pragma unroll
    for (int t = 0; t < 4; ++t) {
      Frag16 b0, b1;
      const half_t* p0 = lK + col * HD + t * 32 + kbB;
      const half_t* p1 = lK + (16 + col) * HD + t * 32 + kbB;
      b0.u[0] = *(const uint4*)p0; b0.u[1] = *(const uint4*)(p0 + 16);
      b1.u[0] = *(const uint4*)p1; b1.u[1] = *(const uint4*)(p1 + 16);
      s0 = wmma_f16(qf[t], b0.h, s0);
      s1 = wmma_f16(qf[t], b1.h, s1);
    }

    // online softmax (rows j+rb per lane; 16-lane reductions stay in half)
    const bool am0 = amask[kc + col] != 0;
    const bool am1 = amask[kc + 16 + col] != 0;
    half_t* pw = lP + wave * (16 * 32);
    float alpha[8];
#pragma unroll
    for (int j = 0; j < 8; ++j) {
      const int qrow = q0 + rb + j;
      float v0 = s0[j], v1 = s1[j];
      if (kc + col > qrow || !am0) v0 = -1e30f;
      if (kc + 16 + col > qrow || !am1) v1 = -1e30f;
      float mx = fmaxf(v0, v1);
#pragma unroll
      for (int off = 8; off >= 1; off >>= 1)
        mx = fmaxf(mx, __shfl_xor(mx, off, 32));
      const float mnew = fmaxf(mrun[j], mx);
      const float a  = __expf(mrun[j] - mnew);
      const float p0 = __expf(v0 - mnew);
      const float p1 = __expf(v1 - mnew);
      float sum = p0 + p1;
#pragma unroll
      for (int off = 8; off >= 1; off >>= 1)
        sum += __shfl_xor(sum, off, 32);
      lrun[j] = lrun[j] * a + sum;
      mrun[j] = mnew;
      alpha[j] = a;
      pw[(rb + j) * 32 + col]      = (half_t)p0;
      pw[(rb + j) * 32 + 16 + col] = (half_t)p1;
    }
#pragma unroll
    for (int t = 0; t < 8; ++t)
#pragma unroll
      for (int j = 0; j < 8; ++j) oacc[t][j] *= alpha[j];

    // P as A-fragment (wave-private LDS round trip, DS in-order per wave)
    Frag16 pf;
    {
      const half_t* pp = pw + col * 32;
      pf.u[0] = *(const uint4*)(pp + kbA);
      pf.u[1] = *(const uint4*)(pp + 16 + kbA);
    }
#pragma unroll
    for (int t = 0; t < 8; ++t) {
      Frag16 vf;
      const half_t* vp = lV + (t * 16 + col) * 32 + kbB;
      vf.u[0] = *(const uint4*)vp;
      vf.u[1] = *(const uint4*)(vp + 16);
      oacc[t] = wmma_f16(pf.h, vf.h, oacc[t]);
    }
  }

  float inv[8];
#pragma unroll
  for (int j = 0; j < 8; ++j) inv[j] = 1.0f / lrun[j];
#pragma unroll
  for (int t = 0; t < 8; ++t)
#pragma unroll
    for (int j = 0; j < 8; ++j)
      O[(long)(q0 + rb + j) * (H * HD) + h * HD + t * 16 + col] =
          (half_t)(oacc[t][j] * inv[j]);
}

// ---------------------------------------------------------------------------
// Host launcher
// ---------------------------------------------------------------------------
extern "C" void kernel_launch(void* const* d_in, const int* in_sizes, int n_in,
                              void* d_out, int out_size, void* d_ws, size_t ws_size,
                              hipStream_t stream) {
  const float* hs   = (const float*)d_in[0];
  const float* cosb = (const float*)d_in[1];
  const float* sinb = (const float*)d_in[2];
  const unsigned char* amask = (const unsigned char*)d_in[3];
  const float* wq = (const float*)d_in[4];
  const float* wk = (const float*)d_in[5];
  const float* wv = (const float*)d_in[6];
  const float* wo = (const float*)d_in[7];
  float* out = (float*)d_out;

  char* ws = (char*)d_ws;
  size_t off = 0;
  auto alloc = [&](size_t bytes) -> char* {
    char* p = ws + off;
    off += (bytes + 255) & ~(size_t)255;
    return p;
  };
  half_t* Xh    = (half_t*)alloc((size_t)S * D * 2);        // hidden f16
  half_t* Wtqkv = (half_t*)alloc((size_t)QKVN * D * 2);     // [6144][4096]
  half_t* Wto   = (half_t*)alloc((size_t)D * D * 2);        // [4096][4096]
  half_t* QKVh  = (half_t*)alloc((size_t)S * QKVN * 2);     // [2048][6144]
  half_t* Qh    = (half_t*)alloc((size_t)H * S * HD * 2);
  half_t* Kh    = (half_t*)alloc((size_t)KVH * S * HD * 2);
  half_t* Vh    = (half_t*)alloc((size_t)KVH * S * HD * 2);
  half_t* AOh   = (half_t*)alloc((size_t)S * H * HD * 2);   // attn out f16

  // 1) casts / transposes
  cast_f32_f16_kernel<<<dim3(((long)S * D) / 1024), 256, 0, stream>>>(hs, Xh, (long)S * D);
  dim3 tb(32, 8);
  transpose_cast_kernel<<<dim3(D / 32, D / 32), tb, 0, stream>>>(wq, Wtqkv, D, D);
  transpose_cast_kernel<<<dim3((KVH * HD) / 32, D / 32), tb, 0, stream>>>(
      wk, Wtqkv + (size_t)(H * HD) * D, D, KVH * HD);
  transpose_cast_kernel<<<dim3((KVH * HD) / 32, D / 32), tb, 0, stream>>>(
      wv, Wtqkv + (size_t)(H * HD + KVH * HD) * D, D, KVH * HD);
  transpose_cast_kernel<<<dim3(D / 32, D / 32), tb, 0, stream>>>(wo, Wto, D, D);

  // 2) fused QKV projection (WMMA)
  gemm_nt_kernel<half_t><<<dim3(S / 128, QKVN / 128), 256, 0, stream>>>(
      Xh, Wtqkv, QKVh, S, QKVN, D);

  // 3) RoPE + head-major relayout (Q pre-scaled by 1/sqrt(HD))
  rope_relayout_kernel<<<dim3(S), 128, 0, stream>>>(QKVh, cosb, sinb, Qh, Kh, Vh);

  // 4) causal flash attention (WMMA)
  flash_attn_kernel<<<dim3(S / 64, H), 128, 0, stream>>>(Qh, Kh, Vh, amask, AOh);

  // 5) output projection (WMMA, f32 out)
  gemm_nt_kernel<float><<<dim3(S / 128, D / 128), 256, 0, stream>>>(
      AOh, Wto, out, S, D, D);
}
